// DeepGCNWithResidual_39238821216994
// MI455X (gfx1250) — matrix operand order repaired and checked by reading, not compiled, not run on hardware
//
#include <hip/hip_runtime.h>

#define D_H 128
#define N_GRAPHS 64
#define LDA 136   // padded LDS row stride (bf16 elements): 272B -> conflict-free b128 loads

typedef __attribute__((ext_vector_type(16))) __bf16 v16bf;
typedef __attribute__((ext_vector_type(8)))  __bf16 v8bf;
typedef __attribute__((ext_vector_type(8)))  float  v8f;

// ---------------------------------------------------------------- degree/norm
__global__ void k_init_deg(float* deg, int N) {
    int i = blockIdx.x * blockDim.x + threadIdx.x;
    if (i < N) deg[i] = 1.0f;                    // self-loop contributes 1
}
__global__ void k_deg_accum(const int* __restrict__ col, float* deg, int E) {
    int e = blockIdx.x * blockDim.x + threadIdx.x;
    if (e < E) atomicAdd(&deg[col[e]], 1.0f);
}
__global__ void k_dinv(float* deg, int N) {
    int i = blockIdx.x * blockDim.x + threadIdx.x;
    if (i < N) { float d = deg[i]; deg[i] = d > 0.f ? rsqrtf(d) : 0.f; }
}
__global__ void k_norm(const int* __restrict__ row, const int* __restrict__ col,
                       const float* __restrict__ dinv, float* norm, int E) {
    int e = blockIdx.x * blockDim.x + threadIdx.x;
    if (e < E) norm[e] = dinv[row[e]] * dinv[col[e]];
}

// ---------------------------------------------------------------- WMMA GEMM
// C[M,128] = A[M,128] @ W[128,128], bf16 inputs, f32 accumulate.
// Fused epilogue: ACC[m,n] = dinv[m]^2 * C[m,n] + bias[n]  (self-loop init for
// the edge scatter). Block = 128 threads = 4 waves; each wave a 16x128 strip
// as 8 16x16 WMMA tiles, K in 4 steps of 32. W is staged TRANSPOSED in LDS so
// both A and B fragments are contiguous 16B runs -> ds_load_b128.
__global__ __launch_bounds__(128)
void k_gemm_wmma(const float* __restrict__ A, const float* __restrict__ W,
                 const float* __restrict__ dinv, const float* __restrict__ bias,
                 float* __restrict__ C, float* __restrict__ ACC, int M)
{
    __shared__ __bf16 sWt[D_H * LDA];  // W transposed: sWt[n*LDA + k]
    __shared__ __bf16 sA [64  * LDA];  // sA[r*LDA + k]
    const int t    = threadIdx.x;
    const int wave = t >> 5;
    const int lane = t & 31;
    const int hf   = lane >> 4;        // which 16-lane half
    const int ln15 = lane & 15;
    const int rowBase = blockIdx.x * 64;

    // stage W transposed (coalesced global read, strided LDS write)
    for (int i = t; i < D_H * D_H; i += 128) {
        int k = i >> 7, n = i & 127;
        sWt[n * LDA + k] = (__bf16)W[i];
    }
    // stage A tile
    for (int i = t; i < 64 * D_H; i += 128) {
        int r = i >> 7, c = i & 127;
        int gr = rowBase + r;
        sA[r * LDA + c] = (__bf16)((gr < M) ? A[(long long)gr * D_H + c] : 0.f);
    }
    __syncthreads();

    const int aRow = wave * 16 + ln15;
    const v8f vzero = {0.f, 0.f, 0.f, 0.f, 0.f, 0.f, 0.f, 0.f};
    v8f acc[8];
#pragma unroll
    for (int nt = 0; nt < 8; ++nt) acc[nt] = vzero;

#pragma unroll
    for (int kb = 0; kb < 4; ++kb) {
        // A fragment (16-bit A 16x32: lane = M, VGPR packs K pairs):
        // elements 0..7  = sA[row][kb*32 + hf*8 .. +7]
        // elements 8..15 = sA[row][kb*32 + 16 + hf*8 .. +7]
        const __bf16* ap = sA + aRow * LDA + kb * 32 + hf * 8;
        v8bf alo = *(const v8bf*)(ap);
        v8bf ahi = *(const v8bf*)(ap + 16);
        v16bf aF = __builtin_shufflevector(alo, ahi,
            0, 1, 2, 3, 4, 5, 6, 7, 8, 9, 10, 11, 12, 13, 14, 15);
#pragma unroll
        for (int nt = 0; nt < 8; ++nt) {
            // B fragment (16-bit B 32x16: lane = N; same K order as A):
            // elements 0..7 = K kb*32 + hf*8 .. +7, elements 8..15 = K kb*32 + 16 + hf*8 .. +7
            const __bf16* bp = sWt + (nt * 16 + ln15) * LDA + kb * 32 + hf * 8;
            v8bf blo = *(const v8bf*)(bp);
            v8bf bhi = *(const v8bf*)(bp + 16);
            v16bf bF = __builtin_shufflevector(blo, bhi,
                0, 1, 2, 3, 4, 5, 6, 7, 8, 9, 10, 11, 12, 13, 14, 15);
            acc[nt] = __builtin_amdgcn_wmma_f32_16x16x32_bf16(
                false, aF, false, bF, (short)0, acc[nt], false, false);
        }
    }

    // D layout: VGPR r holds row (wave*16 + r + 8*hf), col = nt*16 + ln15
#pragma unroll
    for (int r = 0; r < 8; ++r) {
        int gm = rowBase + wave * 16 + r + 8 * hf;
        if (gm < M) {
            float dv = dinv[gm];
            float dv2 = dv * dv;
            long long rb = (long long)gm * D_H;
#pragma unroll
            for (int nt = 0; nt < 8; ++nt) {
                int n = nt * 16 + ln15;
                float val = acc[nt][r];
                C[rb + n]   = val;
                ACC[rb + n] = dv2 * val + bias[n];
            }
        }
    }
}

// One wave per edge: coalesced float4 gather of source row, contiguous f32
// atomics into destination row.
__global__ __launch_bounds__(256)
void k_scatter(const float* __restrict__ H, const int* __restrict__ row,
               const int* __restrict__ col, const float* __restrict__ norm,
               float* __restrict__ OUT, int E)
{
    int e = blockIdx.x * 8 + (threadIdx.x >> 5);
    if (e >= E) return;
    int lane = threadIdx.x & 31;
    int r = row[e], c = col[e];
    float w = norm[e];
    const float4 h4 = *(const float4*)(H + (long long)r * D_H + lane * 4);
    float* o = OUT + (long long)c * D_H + lane * 4;
    atomicAdd(o + 0, w * h4.x);
    atomicAdd(o + 1, w * h4.y);
    atomicAdd(o + 2, w * h4.z);
    atomicAdd(o + 3, w * h4.w);
}

// ReLU + LayerNorm (+ optional residual), one wave (32 lanes x 4 cols) / node.
__global__ __launch_bounds__(128)
void k_relu_ln(const float* __restrict__ IN, const float* __restrict__ gamma,
               const float* __restrict__ beta, const float* __restrict__ RES,
               float* __restrict__ OUT, int N)
{
    int node = blockIdx.x * 4 + (threadIdx.x >> 5);
    if (node >= N) return;
    int lane = threadIdx.x & 31;
    int c = lane * 4;
    long long base = (long long)node * D_H;
    float4 v = *(const float4*)(IN + base + c);
    v.x = fmaxf(v.x, 0.f); v.y = fmaxf(v.y, 0.f);
    v.z = fmaxf(v.z, 0.f); v.w = fmaxf(v.w, 0.f);
    float s = v.x + v.y + v.z + v.w;
#pragma unroll
    for (int off = 16; off > 0; off >>= 1) s += __shfl_xor(s, off, 32);
    float mu = s * (1.0f / 128.0f);
    float dx = v.x - mu, dy = v.y - mu, dz = v.z - mu, dw = v.w - mu;
    float q = dx * dx + dy * dy + dz * dz + dw * dw;
#pragma unroll
    for (int off = 16; off > 0; off >>= 1) q += __shfl_xor(q, off, 32);
    float rstd = rsqrtf(q * (1.0f / 128.0f) + 1e-5f);
    float4 g4 = *(const float4*)(gamma + c);
    float4 b4 = *(const float4*)(beta + c);
    float4 o;
    o.x = dx * rstd * g4.x + b4.x;
    o.y = dy * rstd * g4.y + b4.y;
    o.z = dz * rstd * g4.z + b4.z;
    o.w = dw * rstd * g4.w + b4.w;
    if (RES) {
        float4 r4 = *(const float4*)(RES + base + c);
        o.x += r4.x; o.y += r4.y; o.z += r4.z; o.w += r4.w;
    }
    *(float4*)(OUT + base + c) = o;
}

// ---------------------------------------------------------------- pooling/MLP
__global__ void k_pool_init(float* psum, float* pmax, float* pcnt) {
    int i = blockIdx.x * blockDim.x + threadIdx.x;
    if (i < N_GRAPHS * D_H) { psum[i] = 0.f; pmax[i] = -__builtin_inff(); }
    if (i < N_GRAPHS) pcnt[i] = 0.f;
}

__global__ __launch_bounds__(128)
void k_pool(const float* __restrict__ H, const int* __restrict__ batch,
            float* psum, float* pmax, float* pcnt, int N)
{
    int node = blockIdx.x * 4 + (threadIdx.x >> 5);
    if (node >= N) return;
    int lane = threadIdx.x & 31;
    int g = batch[node];
    long long base = (long long)node * D_H;
    float4 v = *(const float4*)(H + base + lane * 4);
    int c = g * D_H + lane * 4;
    atomicAdd(&psum[c + 0], v.x);
    atomicAdd(&psum[c + 1], v.y);
    atomicAdd(&psum[c + 2], v.z);
    atomicAdd(&psum[c + 3], v.w);
    atomicMax(&pmax[c + 0], v.x);
    atomicMax(&pmax[c + 1], v.y);
    atomicMax(&pmax[c + 2], v.z);
    atomicMax(&pmax[c + 3], v.w);
    if (lane == 0) atomicAdd(&pcnt[g], 1.0f);
}

__global__ __launch_bounds__(128)
void k_mlp1(const float* __restrict__ psum, const float* __restrict__ pmax,
            const float* __restrict__ pcnt, const float* __restrict__ W1,
            const float* __restrict__ b1, float* __restrict__ hidden)
{
    int g = blockIdx.x;            // 64 graphs
    int t = threadIdx.x;           // 128 hidden units
    __shared__ float gv[2 * D_H];
    float cnt = fmaxf(pcnt[g], 1.0f);
    gv[t]       = psum[g * D_H + t] / cnt;
    gv[D_H + t] = pmax[g * D_H + t];
    __syncthreads();
    float a = b1[t];
    for (int k = 0; k < 2 * D_H; ++k) a += gv[k] * W1[k * D_H + t];
    hidden[g * D_H + t] = fmaxf(a, 0.f);
}

__global__ void k_mlp2(const float* __restrict__ hidden, const float* __restrict__ W2,
                       const float* __restrict__ b2, float* __restrict__ out)
{
    int gid = blockIdx.x * blockDim.x + threadIdx.x;
    if (gid >= N_GRAPHS * 3) return;
    int g = gid / 3, o = gid % 3;
    float a = b2[o];
    for (int k = 0; k < D_H; ++k) a += hidden[g * D_H + k] * W2[k * 3 + o];
    out[gid] = a;
}

// ---------------------------------------------------------------- launch
extern "C" void kernel_launch(void* const* d_in, const int* in_sizes, int n_in,
                              void* d_out, int out_size, void* d_ws, size_t ws_size,
                              hipStream_t stream)
{
    const float* x        = (const float*)d_in[0];
    const int*   eidx     = (const int*)d_in[1];
    const int*   batch    = (const int*)d_in[2];
    const float* W_in     = (const float*)d_in[3];
    const float* b_in     = (const float*)d_in[4];
    const float* W_blocks = (const float*)d_in[5];
    const float* b_blocks = (const float*)d_in[6];
    const float* W_out    = (const float*)d_in[7];
    const float* b_out    = (const float*)d_in[8];
    const float* gamma    = (const float*)d_in[9];
    const float* beta     = (const float*)d_in[10];
    const float* fcW1     = (const float*)d_in[11];
    const float* fcb1     = (const float*)d_in[12];
    const float* fcW2     = (const float*)d_in[13];
    const float* fcb2     = (const float*)d_in[14];

    const int N = in_sizes[0] / D_H;
    const int E = in_sizes[1] / 2;
    const int* row = eidx;
    const int* col = eidx + E;

    float* ws    = (float*)d_ws;
    float* dinv  = ws;  ws += N;
    float* norm  = ws;  ws += E;
    float* bufA  = ws;  ws += (size_t)N * D_H;    // current h
    float* bufB  = ws;  ws += (size_t)N * D_H;    // GEMM result (messages)
    float* bufC  = ws;  ws += (size_t)N * D_H;    // conv accumulator
    float* psum  = ws;  ws += N_GRAPHS * D_H;
    float* pmax  = ws;  ws += N_GRAPHS * D_H;
    float* pcnt  = ws;  ws += N_GRAPHS;
    float* hidden = ws; ws += N_GRAPHS * D_H;

    dim3 b256(256), b128(128);
    int gN     = (N + 255) / 256;
    int gE     = (E + 255) / 256;
    int gGemm  = (N + 63) / 64;
    int gNode4 = (N + 3) / 4;
    int gScat  = (E + 7) / 8;

    k_init_deg <<<gN, b256, 0, stream>>>(dinv, N);
    k_deg_accum<<<gE, b256, 0, stream>>>(col, dinv, E);
    k_dinv     <<<gN, b256, 0, stream>>>(dinv, N);
    k_norm     <<<gE, b256, 0, stream>>>(row, col, dinv, norm, E);

    // input conv (GEMM epilogue writes both messages and self-loop-initialized accumulator)
    k_gemm_wmma<<<gGemm, b128, 0, stream>>>(x, W_in, dinv, b_in, bufB, bufC, N);
    k_scatter  <<<gScat, b256, 0, stream>>>(bufB, row, col, norm, bufC, E);
    k_relu_ln  <<<gNode4, b128, 0, stream>>>(bufC, gamma, beta, nullptr, bufA, N);

    // 3 residual blocks
    for (int i = 0; i < 3; ++i) {
        k_gemm_wmma<<<gGemm, b128, 0, stream>>>(bufA, W_blocks + (size_t)i * D_H * D_H,
                                                dinv, b_blocks + (size_t)i * D_H,
                                                bufB, bufC, N);
        k_scatter  <<<gScat, b256, 0, stream>>>(bufB, row, col, norm, bufC, E);
        k_relu_ln  <<<gNode4, b128, 0, stream>>>(bufC, gamma, beta, bufA, bufA, N);
    }

    // output conv (no activation/LN)
    k_gemm_wmma<<<gGemm, b128, 0, stream>>>(bufA, W_out, dinv, b_out, bufB, bufC, N);
    k_scatter  <<<gScat, b256, 0, stream>>>(bufB, row, col, norm, bufC, E);

    // pooling + MLP head
    k_pool_init<<<(N_GRAPHS * D_H + 255) / 256, b256, 0, stream>>>(psum, pmax, pcnt);
    k_pool     <<<gNode4, b128, 0, stream>>>(bufC, batch, psum, pmax, pcnt, N);
    k_mlp1     <<<N_GRAPHS, b128, 0, stream>>>(psum, pmax, pcnt, fcW1, fcb1, hidden);
    k_mlp2     <<<1, 256, 0, stream>>>(hidden, fcW2, fcb2, (float*)d_out);
}
